// DigitConvolutionalModel_38603166057163
// MI455X (gfx1250) — matrix operand order repaired and checked by reading, not compiled, not run on hardware
//
#include <hip/hip_runtime.h>
#include <hip/hip_bf16.h>
#include <cstdint>
#include <cstddef>

// ---------- types ----------
typedef __attribute__((ext_vector_type(16))) __bf16 v16bf;
typedef __attribute__((ext_vector_type(8)))  __bf16 v8bf;
typedef __attribute__((ext_vector_type(8)))  float  v8f;

// ---------- problem sizes (padded for WMMA: K %64==0, N %16==0) ----------
#define B_TOTAL 65536
#define K0 704   // conv features 676 -> 704   (11 * 64)
#define N1 512   // hidden1 500 -> 512
#define N2 256   // hidden2 200 -> 256
#define N3 16    // out 10 -> 16

static __device__ __forceinline__ __bf16 f2bf(float f) {
  // round-to-nearest-even fp32 -> bf16
  unsigned u = __builtin_bit_cast(unsigned, f);
  unsigned r = (u + 0x7FFFu + ((u >> 16) & 1u)) >> 16;
  unsigned short s = (unsigned short)r;
  return __builtin_bit_cast(__bf16, s);
}

// ---------- CDNA5 async global->LDS (ASYNCcnt path) ----------
static __device__ __forceinline__ unsigned lds_off(const void* p) {
  // generic pointer to LDS: addr[31:0] is the LDS byte address (ISA 10.2 aperture table)
  return (unsigned)(uintptr_t)p;
}

static __device__ __forceinline__ void async_ld_b128(unsigned lds, const void* g) {
  // GLOBAL_LOAD_ASYNC_TO_LDS_B128: VDST = LDS byte-address VGPR, VADDR = 64-bit global addr
  asm volatile("global_load_async_to_lds_b128 %0, %1, off"
               :: "v"(lds), "v"((unsigned long long)(uintptr_t)g)
               : "memory");
}

static __device__ __forceinline__ void wait_async0() {
#if __has_builtin(__builtin_amdgcn_s_wait_asynccnt)
  __builtin_amdgcn_s_wait_asynccnt(0);
#else
  asm volatile("s_wait_asynccnt 0x0" ::: "memory");
#endif
}

// ---------- weight prep: W (Kin x Nin row-major) -> Wt (Npad x Kpad bf16, zero padded) ----------
__global__ void prep_wt(const float* __restrict__ W, __bf16* __restrict__ Wt,
                        int Kin, int Nin, int Kpad, int Npad) {
  int idx = blockIdx.x * 256 + threadIdx.x;
  if (idx >= Kpad * Npad) return;
  int n = idx / Kpad;
  int k = idx - n * Kpad;
  float v = (n < Nin && k < Kin) ? W[(size_t)k * Nin + n] : 0.0f;
  Wt[idx] = f2bf(v);
}

__global__ void prep_bias(const float* __restrict__ b, float* __restrict__ bp,
                          int Nin, int Npad) {
  int i = blockIdx.x * 256 + threadIdx.x;
  if (i < Npad) bp[i] = (i < Nin) ? b[i] : 0.0f;
}

// ---------- conv: x (B,784) -> h0 (B, K0) bf16, valid 3x3 cross-correlation ----------
__global__ __launch_bounds__(256) void conv_bf16_kernel(const float* __restrict__ x,
                                                        const float* __restrict__ cwp,
                                                        __bf16* __restrict__ h0) {
  __shared__ __align__(16) float img[4][784];
  const int tid = threadIdx.x;
  const int sub = tid >> 6;    // image slot 0..3
  const int lt  = tid & 63;

  float cw[9];
#pragma unroll
  for (int i = 0; i < 9; ++i) cw[i] = cwp[i];

  for (int i = tid; i < 4 * 784; i += 256) {
    int im = i / 784, off = i - im * 784;
    img[im][off] = x[((size_t)blockIdx.x * 4 + im) * 784 + off];
  }
  __syncthreads();

  const size_t b = (size_t)blockIdx.x * 4 + sub;
  for (int o = lt; o < K0; o += 64) {
    float v = 0.0f;
    if (o < 676) {
      int oi = o / 26, oj = o - oi * 26;
#pragma unroll
      for (int di = 0; di < 3; ++di)
#pragma unroll
        for (int dj = 0; dj < 3; ++dj)
          v += img[sub][(oi + di) * 28 + (oj + dj)] * cw[di * 3 + dj];
    }
    h0[b * K0 + o] = f2bf(v);
  }
}

// ---------- WMMA GEMM: out = act(A[M,K] @ Bt[N,K]^T + bias) ----------
// 256 threads = 8 waves. Block tile 128(M) x NT(N). K-step 64.
// Double-buffered LDS filled by GLOBAL_LOAD_ASYNC_TO_LDS_B128, 1-deep prefetch.
// All fragments for a stage are hoisted into registers before the WMMA burst so
// a single DS wait covers them and the 2*(NT/16) WMMAs issue back-to-back.
template <int NT, bool RELU, bool OUT_BF16>
__global__ __launch_bounds__(256) void gemm_wmma_bf16(
    const __bf16* __restrict__ A, const __bf16* __restrict__ Bt,
    const float* __restrict__ bias, void* __restrict__ outp,
    int K, int ldOut, int Nvalid) {
  constexpr int KS = 64;
  __shared__ __align__(128) __bf16 lA[2][128 * KS];
  __shared__ __align__(128) __bf16 lB[2][NT * KS];

  const int tid  = threadIdx.x;
  const int wave = tid >> 5;
  const int lane = tid & 31;
  const int hh   = lane >> 4;   // lane half
  const int r    = lane & 15;
  const int mBlock = blockIdx.x * 128;
  const int nBlock = blockIdx.y * NT;

  v8f acc[NT / 16];
#pragma unroll
  for (int t = 0; t < NT / 16; ++t) acc[t] = v8f{0.f, 0.f, 0.f, 0.f, 0.f, 0.f, 0.f, 0.f};

  auto issue_tile = [&](int kb, int buf) {
    // A: 128 rows x 64 cols bf16 = 1024 x 16B chunks (8 chunks/row), 4 per thread
#pragma unroll
    for (int i = 0; i < 4; ++i) {
      int idx = tid + i * 256;
      int row = idx >> 3, ch = idx & 7;
      async_ld_b128(lds_off(&lA[buf][row * KS + ch * 8]),
                    A + (size_t)(mBlock + row) * K + kb + ch * 8);
    }
    // B: NT rows x 64 cols = NT*8 chunks
    constexpr int BCH = NT * 8;
#pragma unroll
    for (int i = 0; i < (BCH + 255) / 256; ++i) {
      int idx = tid + i * 256;
      if (idx < BCH) {
        int row = idx >> 3, ch = idx & 7;
        async_ld_b128(lds_off(&lB[buf][row * KS + ch * 8]),
                      Bt + (size_t)(nBlock + row) * K + kb + ch * 8);
      }
    }
  };

  const int nsteps = K / KS;
  issue_tile(0, 0);

  for (int s = 0; s < nsteps; ++s) {
    wait_async0();        // tile s resident in LDS (own wave's async ops)
    __syncthreads();      // ... and everyone else's
    if (s + 1 < nsteps) issue_tile((s + 1) * KS, (s + 1) & 1);

    const __bf16* cA = lA[s & 1];
    const __bf16* cB = lB[s & 1];

    // hoist all fragment loads for this stage into registers
    v16bf afr[2];
    v16bf bfr[2][NT / 16];
#pragma unroll
    for (int ks = 0; ks < 2; ++ks) {
      // A fragment (16x32, interleaved: lanes 0-15 K{0-7,16-23}, lanes 16-31 K{8-15,24-31})
      v8bf alo = *reinterpret_cast<const v8bf*>(&cA[(wave * 16 + r) * KS + ks * 32 + hh * 8]);
      v8bf ahi = *reinterpret_cast<const v8bf*>(&cA[(wave * 16 + r) * KS + ks * 32 + 16 + hh * 8]);
      afr[ks] = __builtin_shufflevector(alo, ahi, 0, 1, 2, 3, 4, 5, 6, 7,
                                        8, 9, 10, 11, 12, 13, 14, 15);
#pragma unroll
      for (int t = 0; t < NT / 16; ++t) {
        // B fragment (32x16): lane = column n; lanes 0-15 K{0-7,16-23}, lanes 16-31 K{8-15,24-31}
        v8bf blo = *reinterpret_cast<const v8bf*>(&cB[(t * 16 + r) * KS + ks * 32 + hh * 8]);
        v8bf bhi = *reinterpret_cast<const v8bf*>(&cB[(t * 16 + r) * KS + ks * 32 + 16 + hh * 8]);
        bfr[ks][t] = __builtin_shufflevector(blo, bhi, 0, 1, 2, 3, 4, 5, 6, 7,
                                             8, 9, 10, 11, 12, 13, 14, 15);
      }
    }

    // WMMA burst: 2*(NT/16) back-to-back matrix ops, disjoint accumulators
#pragma unroll
    for (int ks = 0; ks < 2; ++ks)
#pragma unroll
      for (int t = 0; t < NT / 16; ++t)
        acc[t] = __builtin_amdgcn_wmma_f32_16x16x32_bf16(
            false, afr[ks], false, bfr[ks][t], (short)0, acc[t], false, false);

    __syncthreads();      // compute(s) done before buffer (s&1) is overwritten at s+2
  }

  // epilogue: C/D layout -> VGPR i: lanes 0-15 M=i, lanes 16-31 M=8+i; N=lane%16
#pragma unroll
  for (int t = 0; t < NT / 16; ++t) {
    int ncol = nBlock + t * 16 + r;
    float bv = bias[ncol];
#pragma unroll
    for (int i = 0; i < 8; ++i) {
      int mrow = mBlock + wave * 16 + hh * 8 + i;
      float v = acc[t][i] + bv;
      if (RELU) v = fmaxf(v, 0.0f);
      if (OUT_BF16) {
        reinterpret_cast<__bf16*>(outp)[(size_t)mrow * ldOut + ncol] = f2bf(v);
      } else if (ncol < Nvalid) {
        reinterpret_cast<float*>(outp)[(size_t)mrow * ldOut + ncol] = v;
      }
    }
  }
}

// ---------- workspace layout ----------
static constexpr size_t SZ_H0 = (size_t)B_TOTAL * K0 * 2;
static constexpr size_t SZ_H1 = (size_t)B_TOTAL * N1 * 2;
static constexpr size_t SZ_H2 = (size_t)B_TOTAL * N2 * 2;
static constexpr size_t SZ_W1 = (size_t)N1 * K0 * 2;
static constexpr size_t SZ_W2 = (size_t)N2 * N1 * 2;
static constexpr size_t SZ_W3 = (size_t)N3 * N2 * 2;
static constexpr size_t OFF_H0 = 0;
static constexpr size_t OFF_H1 = OFF_H0 + SZ_H0;
static constexpr size_t OFF_H2 = OFF_H1 + SZ_H1;
static constexpr size_t OFF_W1 = OFF_H2 + SZ_H2;
static constexpr size_t OFF_W2 = OFF_W1 + SZ_W1;
static constexpr size_t OFF_W3 = OFF_W2 + SZ_W2;
static constexpr size_t OFF_B1 = OFF_W3 + SZ_W3;
static constexpr size_t OFF_B2 = OFF_B1 + N1 * 4;
static constexpr size_t OFF_B3 = OFF_B2 + N2 * 4;

extern "C" void kernel_launch(void* const* d_in, const int* in_sizes, int n_in,
                              void* d_out, int out_size, void* d_ws, size_t ws_size,
                              hipStream_t stream) {
  (void)in_sizes; (void)n_in; (void)out_size; (void)ws_size;
  const float* x      = (const float*)d_in[0];
  const float* conv_w = (const float*)d_in[1];
  const float* W1     = (const float*)d_in[2];
  const float* b1     = (const float*)d_in[3];
  const float* W2     = (const float*)d_in[4];
  const float* b2     = (const float*)d_in[5];
  const float* W3     = (const float*)d_in[6];
  const float* b3     = (const float*)d_in[7];
  float* out = (float*)d_out;

  char* ws = (char*)d_ws;
  __bf16* h0  = (__bf16*)(ws + OFF_H0);
  __bf16* h1  = (__bf16*)(ws + OFF_H1);
  __bf16* h2  = (__bf16*)(ws + OFF_H2);
  __bf16* W1t = (__bf16*)(ws + OFF_W1);
  __bf16* W2t = (__bf16*)(ws + OFF_W2);
  __bf16* W3t = (__bf16*)(ws + OFF_W3);
  float*  b1p = (float*)(ws + OFF_B1);
  float*  b2p = (float*)(ws + OFF_B2);
  float*  b3p = (float*)(ws + OFF_B3);

  // weight / bias prep (tiny)
  prep_wt<<<(K0 * N1 + 255) / 256, 256, 0, stream>>>(W1, W1t, 676, 500, K0, N1);
  prep_wt<<<(N1 * N2 + 255) / 256, 256, 0, stream>>>(W2, W2t, 500, 200, N1, N2);
  prep_wt<<<(N2 * N3 + 255) / 256, 256, 0, stream>>>(W3, W3t, 200, 10, N2, N3);
  prep_bias<<<(N1 + 255) / 256, 256, 0, stream>>>(b1, b1p, 500, N1);
  prep_bias<<<(N2 + 255) / 256, 256, 0, stream>>>(b2, b2p, 200, N2);
  prep_bias<<<(N3 + 255) / 256, 256, 0, stream>>>(b3, b3p, 10, N3);

  // conv -> bf16 features (4 images per block)
  conv_bf16_kernel<<<B_TOTAL / 4, 256, 0, stream>>>(x, conv_w, h0);

  // h1 = relu(h0 @ W1 + b1)
  gemm_wmma_bf16<64, true, true><<<dim3(B_TOTAL / 128, N1 / 64), 256, 0, stream>>>(
      h0, W1t, b1p, h1, K0, N1, N1);
  // h2 = relu(h1 @ W2 + b2)
  gemm_wmma_bf16<64, true, true><<<dim3(B_TOTAL / 128, N2 / 64), 256, 0, stream>>>(
      h1, W2t, b2p, h2, N1, N2, N2);
  // out = h2 @ W3 + b3   (fp32, only first 10 columns)
  gemm_wmma_bf16<16, false, false><<<dim3(B_TOTAL / 128, 1), 256, 0, stream>>>(
      h2, W3t, b3p, out, N2, 10, 10);
}
